// MoeRegBlock_15831249453472
// MI455X (gfx1250) — matrix operand order repaired and checked by reading, not compiled, not run on hardware
//
#include <hip/hip_runtime.h>

typedef __attribute__((ext_vector_type(16))) _Float16 v16h;
typedef __attribute__((ext_vector_type(8)))  _Float16 v8h;
typedef __attribute__((ext_vector_type(8)))  float    v8f;
typedef __attribute__((vector_size(16)))     int      gv4i;  // int4 as the async builtin expects

#define DEV static __device__ __forceinline__

DEV v8f vzero8f() { v8f r; for (int i = 0; i < 8; ++i) r[i] = 0.f; return r; }
DEV v8h vzero8h() { v8h r; for (int i = 0; i < 8; ++i) r[i] = (_Float16)0.f; return r; }

// ---- CDNA5 async global->LDS staging (ASYNCcnt pipe), probe-verified present ----
#if defined(__has_builtin)
#if __has_builtin(__builtin_amdgcn_global_load_async_to_lds_b128)
#define HAVE_ASYNC_LDS 1
#endif
#endif

#ifdef HAVE_ASYNC_LDS
#define GLOAD_ASYNC(gp, lp)                                                    \
  __builtin_amdgcn_global_load_async_to_lds_b128(                              \
      (gv4i*)const_cast<void*>((const void*)(gp)),                             \
      (gv4i*)(void*)(lp), 0, 0)
#if __has_builtin(__builtin_amdgcn_s_wait_asynccnt)
#define WAIT_ASYNC() __builtin_amdgcn_s_wait_asynccnt(0)
#else
#define WAIT_ASYNC() asm volatile("s_wait_asynccnt 0" ::: "memory")
#endif
#else
#define WAIT_ASYNC()
#endif

// ============================ f32 -> f16 convert ============================
__global__ void k_cvt_f32_f16(const float* __restrict__ in, _Float16* __restrict__ out, size_t n) {
  size_t i = (size_t)blockIdx.x * blockDim.x + threadIdx.x;
  if (i < n) out[i] = (_Float16)in[i];
}

// f32 (z,K,N) row-major  ->  f16 (z,N,K)  (transposed weights for WMMA B path)
__global__ void k_cvt_tr(const float* __restrict__ in, _Float16* __restrict__ out, int K, int N) {
  size_t base = (size_t)blockIdx.z * K * N;
  size_t i = (size_t)blockIdx.x * blockDim.x + threadIdx.x;
  if (i >= (size_t)K * N) return;
  int k = (int)(i / N), n = (int)(i % N);
  out[base + (size_t)n * K + k] = (_Float16)in[base + i];
}

// ============================ RMSNorm (D=512) ============================
__global__ __launch_bounds__(256) void k_rmsnorm(const float* __restrict__ x,
                                                 const float* __restrict__ w,
                                                 _Float16* __restrict__ out) {
  int t = blockIdx.x;
  int tid = threadIdx.x;
  float a0 = x[(size_t)t * 512 + tid];
  float a1 = x[(size_t)t * 512 + tid + 256];
  float ss = a0 * a0 + a1 * a1;
  for (int off = 16; off >= 1; off >>= 1) ss += __shfl_xor(ss, off, 32);
  __shared__ float sr[8];
  if ((tid & 31) == 0) sr[tid >> 5] = ss;
  __syncthreads();
  float tot = 0.f;
  for (int i = 0; i < 8; ++i) tot += sr[i];
  float sc = rsqrtf(tot * (1.f / 512.f) + 1e-5f);
  out[(size_t)t * 512 + tid]       = (_Float16)(a0 * sc * w[tid]);
  out[(size_t)t * 512 + tid + 256] = (_Float16)(a1 * sc * w[tid + 256]);
}

// ============================ Generic WMMA GEMM ============================
// C[M,N] = op( A[M,K] @ Bt[N,K]^T )  A f16 row-major, Bt f16 TRANSPOSED (N,K).
// 256 thr = 8 waves, block tile 128x128, wave tile 32x64 (2x4 WMMA accum), K step 32.
// Double-buffered LDS; async global->LDS overlaps next tile with current WMMAs.
__global__ __launch_bounds__(256) void k_gemm(
    const _Float16* __restrict__ A, const _Float16* __restrict__ Bt,
    float* __restrict__ Cf, _Float16* __restrict__ Ch,
    int M, int N, int Kd, int lda, int ldbT, int ldc,
    size_t strideB,
    const float* __restrict__ residual,
    const int* __restrict__ gidx, const int* __restrict__ sidx, int listStride,
    const int* __restrict__ cntArr,
    const float* __restrict__ rowScale,
    int gelu) {
  (void)N;
  int ez = blockIdx.z;
  const _Float16* Bz = Bt + (size_t)ez * strideB;
  const int* gl = gidx ? gidx + (size_t)ez * listStride : nullptr;
  const int* sl = sidx ? sidx + (size_t)ez * listStride : nullptr;
  int Meff = cntArr ? cntArr[ez] : M;
  int m0 = blockIdx.y * 128;
  int n0 = blockIdx.x * 128;
  if (m0 >= Meff) return;

  __shared__ __attribute__((aligned(16))) _Float16 As[2][128][40];   // (m, k) + pad
  __shared__ __attribute__((aligned(16))) _Float16 Bts[2][128][40];  // (n, k) + pad

  int tid = threadIdx.x;
  int w = tid >> 5, lane = tid & 31, g = lane >> 4, ln = lane & 15;
  int wm = w >> 1, wn = w & 1;

  // per-thread staging coordinates: 2 chunks of 8 halves for A and for Bt
  int sRow[2], sCol[2], aIdx[2];
  bool aOk[2];
  for (int i = 0; i < 2; ++i) {
    int chunk = tid * 2 + i;
    sRow[i] = chunk >> 2;
    sCol[i] = (chunk & 3) * 8;
    int mrow = m0 + sRow[i];
    aOk[i] = mrow < Meff;
    aIdx[i] = aOk[i] ? (gl ? gl[mrow] : mrow) : 0;
  }

  // gather-padded rows: zero once, never touched by the staging loads below
  for (int i = 0; i < 2; ++i)
    if (!aOk[i]) {
      *(v8h*)(&As[0][sRow[i]][sCol[i]]) = vzero8h();
      *(v8h*)(&As[1][sRow[i]][sCol[i]]) = vzero8h();
    }

  auto issueTile = [&](int buf, int k0) {
    for (int i = 0; i < 2; ++i) {
      if (aOk[i]) {
#ifdef HAVE_ASYNC_LDS
        GLOAD_ASYNC(A + (size_t)aIdx[i] * lda + k0 + sCol[i], &As[buf][sRow[i]][sCol[i]]);
#else
        *(v8h*)(&As[buf][sRow[i]][sCol[i]]) =
            *(const v8h*)(A + (size_t)aIdx[i] * lda + k0 + sCol[i]);
#endif
      }
#ifdef HAVE_ASYNC_LDS
      GLOAD_ASYNC(Bz + (size_t)(n0 + sRow[i]) * ldbT + k0 + sCol[i],
                  &Bts[buf][sRow[i]][sCol[i]]);
#else
      *(v8h*)(&Bts[buf][sRow[i]][sCol[i]]) =
          *(const v8h*)(Bz + (size_t)(n0 + sRow[i]) * ldbT + k0 + sCol[i]);
#endif
    }
  };

  v8f acc[2][4];
  for (int i = 0; i < 2; ++i)
    for (int j = 0; j < 4; ++j) acc[i][j] = vzero8f();

  int nSteps = Kd >> 5;
  issueTile(0, 0);
  for (int step = 0; step < nSteps; ++step) {
    int p = step & 1;
    WAIT_ASYNC();
    __syncthreads();  // buf[p] ready; all readers of buf[p^1] are done
    if (step + 1 < nSteps) issueTile(p ^ 1, (step + 1) << 5);

    // A fragment: lane = M%16, half g, K runs of 8 at {g*8, 16+g*8}
    v16h aF[2];
    for (int ms = 0; ms < 2; ++ms) {
      int row = wm * 32 + ms * 16 + ln;
      v8h lo = *(const v8h*)(&As[p][row][g * 8]);
      v8h hi = *(const v8h*)(&As[p][row][16 + g * 8]);
      for (int j = 0; j < 8; ++j) { aF[ms][j] = lo[j]; aF[ms][j + 8] = hi[j]; }
    }
    // B fragment: lane = N%16, K runs of 8 at {g*8, 16+g*8}
    v16h bF[4];
    for (int ns = 0; ns < 4; ++ns) {
      int col = wn * 64 + ns * 16 + ln;
      v8h lo = *(const v8h*)(&Bts[p][col][g * 8]);
      v8h hi = *(const v8h*)(&Bts[p][col][16 + g * 8]);
      for (int j = 0; j < 8; ++j) { bF[ns][j] = lo[j]; bF[ns][j + 8] = hi[j]; }
    }
    for (int ms = 0; ms < 2; ++ms)
      for (int ns = 0; ns < 4; ++ns)
        acc[ms][ns] = __builtin_amdgcn_wmma_f32_16x16x32_f16(
            false, aF[ms], false, bF[ns], (short)0, acc[ms][ns], false, false);
  }

  // epilogue: C fragment lane = N%16, rows (lane/16)*8+v
  for (int ms = 0; ms < 2; ++ms)
    for (int ns = 0; ns < 4; ++ns)
      for (int v = 0; v < 8; ++v) {
        int rl = wm * 32 + ms * 16 + g * 8 + v;
        int mrow = m0 + rl;
        if (mrow >= Meff) continue;
        int ccol = n0 + wn * 64 + ns * 16 + ln;
        int orow = sl ? sl[mrow] : mrow;
        float val = acc[ms][ns][v];
        if (gelu) val = 0.5f * val * (1.f + erff(val * 0.70710678118f));
        if (rowScale) val *= rowScale[orow];
        if (residual) val += residual[(size_t)orow * ldc + ccol];
        if (Cf) Cf[(size_t)orow * ldc + ccol] = val;
        else    Ch[(size_t)orow * ldc + ccol] = (_Float16)val;
      }
}

// ============================ RoPE + relayout ============================
// q0/k0/v0: (B*S, H*64) f16 -> qh/kh: (B,H,S,64) ; vh: (B,H,64,S) TRANSPOSED
__global__ void k_rope(const _Float16* __restrict__ q0, const _Float16* __restrict__ k0,
                       const _Float16* __restrict__ v0,
                       _Float16* __restrict__ qh, _Float16* __restrict__ kh,
                       _Float16* __restrict__ vh, int Bb, int Ss, int nH) {
  size_t idx = (size_t)blockIdx.x * blockDim.x + threadIdx.x;
  size_t total = (size_t)Bb * Ss * nH * 32;
  if (idx >= total) return;
  int i = (int)(idx & 31);
  size_t r = idx >> 5;
  int hh = (int)(r % nH); r /= nH;
  int s = (int)(r % Ss);
  int b = (int)(r / Ss);
  float freq = powf(10000.f, -(2.f * i) / 64.f);
  float ang = (float)s * freq;
  float c = cosf(ang), sn = sinf(ang);
  size_t src  = ((size_t)b * Ss + s) * 512 + hh * 64;
  size_t dst  = (((size_t)b * nH + hh) * Ss + s) * 64;
  size_t dstT = (((size_t)b * nH + hh) * 64) * Ss;
  float q1 = (float)q0[src + 2 * i], q2 = (float)q0[src + 2 * i + 1];
  qh[dst + 2 * i]     = (_Float16)(q1 * c - q2 * sn);
  qh[dst + 2 * i + 1] = (_Float16)(q1 * sn + q2 * c);
  float k1 = (float)k0[src + 2 * i], k2 = (float)k0[src + 2 * i + 1];
  kh[dst + 2 * i]     = (_Float16)(k1 * c - k2 * sn);
  kh[dst + 2 * i + 1] = (_Float16)(k1 * sn + k2 * c);
  vh[dstT + (size_t)(2 * i) * Ss + s]     = v0[src + 2 * i];
  vh[dstT + (size_t)(2 * i + 1) * Ss + s] = v0[src + 2 * i + 1];
}

// ============================ Flash attention (causal, dh=64) ============================
// grid (S/64, H, B), 128 thr = 4 waves; wave owns 16 query rows.
// All fragment loads are 128-bit vector ops (V consumed transposed).
__global__ __launch_bounds__(128) void k_flash(const _Float16* __restrict__ Q,
                                               const _Float16* __restrict__ K,
                                               const _Float16* __restrict__ Vt,
                                               _Float16* __restrict__ Oh,
                                               int Ss, int nH, int Dm) {
  int b = blockIdx.z, hId = blockIdx.y, qt = blockIdx.x;
  int tid = threadIdx.x;
  int w = tid >> 5, lane = tid & 31, g = lane >> 4, ln = lane & 15;
  int qbase = qt * 64 + w * 16;
  const size_t headOff = ((size_t)(b * nH + hId)) * Ss * 64;
  const _Float16* Qp = Q + headOff;
  const _Float16* Kp = K + headOff;
  const _Float16* Vp = Vt + headOff;  // (64, Ss) per head

  // Q A-fragments, pre-scaled by 1/sqrt(64): two contiguous 8-half runs per kc
  v16h aQ[2];
  {
    const _Float16* qrow = Qp + (size_t)(qbase + ln) * 64;
    for (int kc = 0; kc < 2; ++kc) {
      v8h lo = *(const v8h*)(qrow + kc * 32 + g * 8);
      v8h hi = *(const v8h*)(qrow + kc * 32 + 16 + g * 8);
      for (int j = 0; j < 8; ++j) {
        aQ[kc][j]     = (_Float16)((float)lo[j] * 0.125f);
        aQ[kc][j + 8] = (_Float16)((float)hi[j] * 0.125f);
      }
    }
  }

  float m[8], l[8];
  v8f o[4];
  for (int v = 0; v < 8; ++v) { m[v] = -1e30f; l[v] = 0.f; }
  for (int n = 0; n < 4; ++n) o[n] = vzero8f();

  __shared__ __attribute__((aligned(16))) _Float16 Pl[4][16 * 32];

  int qTileEnd = qt * 64 + 63;  // uniform across block -> safe __syncthreads
  for (int j0 = 0; j0 <= qTileEnd; j0 += 32) {
    v8f s[2];
    for (int nc = 0; nc < 2; ++nc) {
      int key = j0 + nc * 16 + ln;
      const _Float16* krow = Kp + (size_t)key * 64 + g * 8;
      v8h k0lo = *(const v8h*)(krow);
      v8h k0hi = *(const v8h*)(krow + 16);
      v8h k1lo = *(const v8h*)(krow + 32);
      v8h k1hi = *(const v8h*)(krow + 48);
      v16h bK0, bK1;
      for (int j = 0; j < 8; ++j) {
        bK0[j] = k0lo[j]; bK0[j + 8] = k0hi[j];
        bK1[j] = k1lo[j]; bK1[j + 8] = k1hi[j];
      }
      v8f accs = vzero8f();
      accs = __builtin_amdgcn_wmma_f32_16x16x32_f16(false, aQ[0], false, bK0, (short)0, accs, false, false);
      accs = __builtin_amdgcn_wmma_f32_16x16x32_f16(false, aQ[1], false, bK1, (short)0, accs, false, false);
      s[nc] = accs;
    }
    // causal mask + online softmax (row = qbase+g*8+v, col = j0+nc*16+ln)
    float rmax[8];
    for (int v = 0; v < 8; ++v) {
      int row = qbase + g * 8 + v;
      for (int nc = 0; nc < 2; ++nc) {
        int key = j0 + nc * 16 + ln;
        if (key > row) s[nc][v] = -1e30f;
      }
      rmax[v] = fmaxf(s[0][v], s[1][v]);
    }
    for (int off = 8; off >= 1; off >>= 1)
      for (int v = 0; v < 8; ++v) rmax[v] = fmaxf(rmax[v], __shfl_xor(rmax[v], off, 32));
    float scl[8];
    for (int v = 0; v < 8; ++v) {
      float mn = fmaxf(m[v], rmax[v]);
      scl[v] = __expf(m[v] - mn);
      m[v] = mn;
    }
    float rsum[8];
    for (int v = 0; v < 8; ++v) {
      float p0 = __expf(s[0][v] - m[v]);
      float p1 = __expf(s[1][v] - m[v]);
      s[0][v] = p0; s[1][v] = p1;
      rsum[v] = p0 + p1;
    }
    for (int off = 8; off >= 1; off >>= 1)
      for (int v = 0; v < 8; ++v) rsum[v] += __shfl_xor(rsum[v], off, 32);
    for (int v = 0; v < 8; ++v) l[v] = l[v] * scl[v] + rsum[v];
    for (int n = 0; n < 4; ++n)
      for (int v = 0; v < 8; ++v) o[n][v] *= scl[v];
    // relay P: C-fragment -> LDS (16x32 row-major) -> A-fragment (vectorized)
    for (int nc = 0; nc < 2; ++nc)
      for (int v = 0; v < 8; ++v)
        Pl[w][(g * 8 + v) * 32 + nc * 16 + ln] = (_Float16)s[nc][v];
    __syncthreads();
    v16h pA;
    {
      v8h plo = *(const v8h*)(&Pl[w][ln * 32 + g * 8]);
      v8h phi = *(const v8h*)(&Pl[w][ln * 32 + 16 + g * 8]);
      for (int j = 0; j < 8; ++j) { pA[j] = plo[j]; pA[j + 8] = phi[j]; }
    }
    // O += P @ V   (V transposed: contiguous along key per dh column)
    for (int n = 0; n < 4; ++n) {
      const _Float16* vcol = Vp + (size_t)(n * 16 + ln) * Ss + j0 + g * 8;
      v8h vlo = *(const v8h*)(vcol);
      v8h vhi = *(const v8h*)(vcol + 16);
      v16h bV;
      for (int j = 0; j < 8; ++j) { bV[j] = vlo[j]; bV[j + 8] = vhi[j]; }
      o[n] = __builtin_amdgcn_wmma_f32_16x16x32_f16(false, pA, false, bV, (short)0, o[n], false, false);
    }
    __syncthreads();
  }
  // normalize; store (T, D) f16 for output projection
  for (int v = 0; v < 8; ++v) {
    int row = qbase + g * 8 + v;
    float inv = 1.f / l[v];
    size_t t = (size_t)b * Ss + row;
    for (int n = 0; n < 4; ++n)
      Oh[t * Dm + hId * 64 + n * 16 + ln] = (_Float16)(o[n][v] * inv);
  }
}

// ============================ Router: softmax + top-2 + gather lists ============================
__global__ __launch_bounds__(256) void k_router(const _Float16* __restrict__ hn,
                                                const float* __restrict__ rw,
                                                float* __restrict__ gate, int* __restrict__ cnt,
                                                int* __restrict__ lst_tok, int* __restrict__ lst_slot,
                                                int T) {
  int t = blockIdx.x;
  int tid = threadIdx.x;
  float p[8];
  for (int e = 0; e < 8; ++e) p[e] = 0.f;
  for (int d = tid; d < 512; d += 256) {
    float xv = (float)hn[(size_t)t * 512 + d];
    for (int e = 0; e < 8; ++e) p[e] += xv * rw[d * 8 + e];
  }
  for (int off = 16; off >= 1; off >>= 1)
    for (int e = 0; e < 8; ++e) p[e] += __shfl_xor(p[e], off, 32);
  __shared__ float sm[8][8];
  if ((tid & 31) == 0)
    for (int e = 0; e < 8; ++e) sm[tid >> 5][e] = p[e];
  __syncthreads();
  if (tid == 0) {
    float lg[8];
    for (int e = 0; e < 8; ++e) {
      float a = 0.f;
      for (int wv = 0; wv < 8; ++wv) a += sm[wv][e];
      lg[e] = a;
    }
    float mx = lg[0];
    for (int e = 1; e < 8; ++e) mx = fmaxf(mx, lg[e]);
    float pe[8];
    for (int e = 0; e < 8; ++e) pe[e] = __expf(lg[e] - mx);
    int i0 = 0;
    for (int e = 1; e < 8; ++e) if (pe[e] > pe[i0]) i0 = e;
    int i1 = (i0 == 0) ? 1 : 0;
    for (int e = 0; e < 8; ++e) if (e != i0 && pe[e] > pe[i1]) i1 = e;
    float s2 = pe[i0] + pe[i1];
    gate[t * 2]     = pe[i0] / s2;
    gate[t * 2 + 1] = pe[i1] / s2;
    int p0 = atomicAdd(&cnt[i0], 1);
    lst_tok[i0 * T + p0] = t;
    lst_slot[i0 * T + p0] = t * 2;
    int p1 = atomicAdd(&cnt[i1], 1);
    lst_tok[i1 * T + p1] = t;
    lst_slot[i1 * T + p1] = t * 2 + 1;
  }
}

// ============================ final: out = h + slot0 + slot1 ============================
__global__ void k_final(const float* __restrict__ h, const float* __restrict__ slot,
                        float* __restrict__ out, size_t n) {
  size_t i = (size_t)blockIdx.x * blockDim.x + threadIdx.x;
  if (i >= n) return;
  size_t t = i >> 9;
  int d = (int)(i & 511);
  out[i] = h[i] + slot[(t * 2) * 512 + d] + slot[(t * 2 + 1) * 512 + d];
}

// ============================ launch ============================
extern "C" void kernel_launch(void* const* d_in, const int* in_sizes, int n_in,
                              void* d_out, int out_size, void* d_ws, size_t ws_size,
                              hipStream_t stream) {
  (void)in_sizes; (void)n_in; (void)out_size; (void)ws_size;
  constexpr int Bb = 4, Ss = 2048, Dm = 512, nH = 8, Ee = 8, Hf = 2048;
  constexpr int T = Bb * Ss;

  const float* x       = (const float*)d_in[0];
  const float* attn_nw = (const float*)d_in[1];
  const float* wq      = (const float*)d_in[2];
  const float* wk      = (const float*)d_in[3];
  const float* wv      = (const float*)d_in[4];
  const float* wo      = (const float*)d_in[5];
  const float* ffn_nw  = (const float*)d_in[6];
  const float* rw      = (const float*)d_in[7];
  const float* w1      = (const float*)d_in[8];
  const float* w2      = (const float*)d_in[9];

  char* wsp = (char*)d_ws;
  size_t off = 0;
  auto alloc = [&](size_t bytes) -> void* {
    void* p = wsp + off;
    off += (bytes + 255) & ~(size_t)255;
    return p;
  };

  _Float16* wq_t   = (_Float16*)alloc((size_t)Dm * Dm * 2);
  _Float16* wk_t   = (_Float16*)alloc((size_t)Dm * Dm * 2);
  _Float16* wv_t   = (_Float16*)alloc((size_t)Dm * Dm * 2);
  _Float16* wo_t   = (_Float16*)alloc((size_t)Dm * Dm * 2);
  _Float16* w1_t   = (_Float16*)alloc((size_t)Ee * Dm * Hf * 2);
  _Float16* w2_t   = (_Float16*)alloc((size_t)Ee * Hf * Dm * 2);
  _Float16* xn_h   = (_Float16*)alloc((size_t)T * Dm * 2);
  _Float16* q0     = (_Float16*)alloc((size_t)T * Dm * 2);
  _Float16* k0     = (_Float16*)alloc((size_t)T * Dm * 2);
  _Float16* v0     = (_Float16*)alloc((size_t)T * Dm * 2);
  _Float16* qh     = (_Float16*)alloc((size_t)T * Dm * 2);
  _Float16* kh     = (_Float16*)alloc((size_t)T * Dm * 2);
  _Float16* vh     = (_Float16*)alloc((size_t)T * Dm * 2);
  _Float16* attn_h = (_Float16*)alloc((size_t)T * Dm * 2);
  float*    hres   = (float*)   alloc((size_t)T * Dm * 4);
  _Float16* hn_h   = (_Float16*)alloc((size_t)T * Dm * 2);
  float*    gate   = (float*)   alloc((size_t)2 * T * 4);
  int*      lst_tok  = (int*)   alloc((size_t)Ee * T * 4);
  int*      lst_slot = (int*)   alloc((size_t)Ee * T * 4);
  int*      cnt    = (int*)     alloc((size_t)Ee * 4);
  _Float16* hid_h  = (_Float16*)alloc((size_t)2 * T * Hf * 2);
  float*    slotout = (float*)  alloc((size_t)2 * T * Dm * 4);

  (void)hipMemsetAsync(cnt, 0, Ee * sizeof(int), stream);

  // transposed f16 weights (one-time per launch)
  {
    unsigned nblk = (unsigned)(((size_t)Dm * Dm + 255) / 256);
    dim3 g1(nblk, 1, 1);
    k_cvt_tr<<<g1, 256, 0, stream>>>(wq, wq_t, Dm, Dm);
    k_cvt_tr<<<g1, 256, 0, stream>>>(wk, wk_t, Dm, Dm);
    k_cvt_tr<<<g1, 256, 0, stream>>>(wv, wv_t, Dm, Dm);
    k_cvt_tr<<<g1, 256, 0, stream>>>(wo, wo_t, Dm, Dm);
    dim3 g2((unsigned)(((size_t)Dm * Hf + 255) / 256), 1, Ee);
    k_cvt_tr<<<g2, 256, 0, stream>>>(w1, w1_t, Dm, Hf);
    k_cvt_tr<<<g2, 256, 0, stream>>>(w2, w2_t, Hf, Dm);
  }

  // xn = rmsnorm(x)
  k_rmsnorm<<<T, 256, 0, stream>>>(x, attn_nw, xn_h);

  // QKV projections
  dim3 gqkv(Dm / 128, T / 128, 1);
  k_gemm<<<gqkv, 256, 0, stream>>>(xn_h, wq_t, nullptr, q0, T, Dm, Dm, Dm, Dm, Dm,
                                   (size_t)0, nullptr, nullptr, nullptr, 0, nullptr, nullptr, 0);
  k_gemm<<<gqkv, 256, 0, stream>>>(xn_h, wk_t, nullptr, k0, T, Dm, Dm, Dm, Dm, Dm,
                                   (size_t)0, nullptr, nullptr, nullptr, 0, nullptr, nullptr, 0);
  k_gemm<<<gqkv, 256, 0, stream>>>(xn_h, wv_t, nullptr, v0, T, Dm, Dm, Dm, Dm, Dm,
                                   (size_t)0, nullptr, nullptr, nullptr, 0, nullptr, nullptr, 0);

  // RoPE + relayout (V transposed per head)
  size_t nrope = (size_t)T * nH * 32;
  k_rope<<<(unsigned)((nrope + 255) / 256), 256, 0, stream>>>(q0, k0, v0, qh, kh, vh, Bb, Ss, nH);

  // flash attention
  dim3 gfa(Ss / 64, nH, Bb);
  k_flash<<<gfa, 128, 0, stream>>>(qh, kh, vh, attn_h, Ss, nH, Dm);

  // output projection + residual: hres = x + attn @ wo
  k_gemm<<<gqkv, 256, 0, stream>>>(attn_h, wo_t, hres, nullptr, T, Dm, Dm, Dm, Dm, Dm,
                                   (size_t)0, x, nullptr, nullptr, 0, nullptr, nullptr, 0);

  // hn = rmsnorm(hres); router top-2
  k_rmsnorm<<<T, 256, 0, stream>>>(hres, ffn_nw, hn_h);
  k_router<<<T, 256, 0, stream>>>(hn_h, rw, gate, cnt, lst_tok, lst_slot, T);

  // MoE up: hid[slot] = gelu(hn[token] @ w1[e])
  dim3 gm1(Hf / 128, T / 128, Ee);
  k_gemm<<<gm1, 256, 0, stream>>>(hn_h, w1_t, nullptr, hid_h, T, Hf, Dm, Dm, Dm, Hf,
                                  (size_t)Dm * Hf, nullptr, lst_tok, lst_slot, T, cnt, nullptr, 1);

  // MoE down: slotout[slot] = gate[slot] * (hid[slot] @ w2[e])
  dim3 gm2(Dm / 128, T / 128, Ee);
  k_gemm<<<gm2, 256, 0, stream>>>(hid_h, w2_t, slotout, nullptr, T, Dm, Hf, Hf, Hf, Dm,
                                  (size_t)Hf * Dm, nullptr, lst_slot, lst_slot, T, cnt, gate, 0);

  // out = hres + slot0 + slot1
  size_t ntot = (size_t)T * Dm;
  k_final<<<(unsigned)((ntot + 255) / 256), 256, 0, stream>>>(hres, slotout, (float*)d_out, ntot);
}
